// VoxelTracker_30356828848307
// MI455X (gfx1250) — matrix operand order repaired and checked by reading, not compiled, not run on hardware
//
#include <hip/hip_runtime.h>
#include <hip/hip_bf16.h>
#include <math.h>

// ---- problem constants (from reference) ----
#define N_   2
#define C_   32
#define X_   48
#define Y_   48
#define Z_   24
#define SPAT (X_*Y_*Z_)      // 55296
#define VOX  (N_*SPAT)       // 110592

typedef _Float16 v16h __attribute__((ext_vector_type(16)));
typedef _Float16 v8h  __attribute__((ext_vector_type(8)));
typedef float    v8f  __attribute__((ext_vector_type(8)));

// Wave-local LDS ordering: all 32 lanes of a wave have issued their ds_stores
// by the time this instruction executes (lockstep issue); s_wait_dscnt 0 makes
// the data visible, and the "memory" clobber stops the compiler from hoisting
// the subsequent LDS reads above it. No block barrier needed: each wave only
// touches its own LDS slice.
__device__ __forceinline__ void wave_lds_fence() {
    asm volatile("s_wait_dscnt 0x0" ::: "memory");
}

// ============================================================================
// Pass 1: L2-normalize over channels, convert to f16, store channel-last:
//   ws layout: fnorm[2][VOX][32] f16  (f0n first, f1n second) = 14.2 MB
//   Reads are coalesced (consecutive threads -> consecutive spatial index),
//   writes are 64B-contiguous per thread.
// ============================================================================
__global__ __launch_bounds__(256)
void vt_normalize_kernel(const float* __restrict__ f0,
                         const float* __restrict__ f1,
                         _Float16* __restrict__ outp)
{
    int tid = blockIdx.x * blockDim.x + threadIdx.x;
    if (tid >= 2 * VOX) return;
    const float* src = (tid >= VOX) ? f1 : f0;
    int v = (tid >= VOX) ? (tid - VOX) : tid;
    int n = v / SPAT, s = v - n * SPAT;

    float vals[32];
    float ss = 0.0f;
#pragma unroll
    for (int c = 0; c < 32; ++c) {
        float x = src[(size_t)(n * C_ + c) * SPAT + s];
        vals[c] = x;
        ss += x * x;
    }
    float sc = 1.0f / (sqrtf(ss) + 1e-7f);
    _Float16* dst = outp + (size_t)tid * 32;
#pragma unroll
    for (int c = 0; c < 32; ++c)
        dst[c] = (_Float16)(vals[c] * sc);
}

// ============================================================================
// Pass 2: cost volume via WMMA diagonal-band trick + streaming softmax.
// One wave (32 lanes) per (n, x, y, z-tile). Two z-tiles cover Z=24:
//   tile 0: z0=0,  output rows 0..13  (z = 0..13)
//   tile 1: z0=10, output rows 4..13  (z = 14..23)
// For each of 9 (dx,dy): B columns j hold f1n at z' = z0-1+j (zeros when the
// neighbor is out of grid -> dot = 0 -> logit = 0, exactly as the reference's
// valid-mask). D[i, i+1+dz] is the logit for offset (dx,dy,dz).
// ============================================================================
__global__ __launch_bounds__(128)
void vt_costvol_flow_kernel(const _Float16* __restrict__ fnorm,
                            const float* __restrict__ logit_scale,
                            const int* __restrict__ training,
                            float* __restrict__ out)
{
    __shared__ float tile[4][16 * 16];

    const int lane = threadIdx.x & 31;
    const int wid  = threadIdx.x >> 5;
    int u = blockIdx.x * 4 + wid;
    const int t = u & 1;  u >>= 1;
    const int y = u % Y_; u /= Y_;
    const int x = u % X_;
    const int n = u / X_;

    const int z0  = t ? 10 : 0;
    const int rlo = t ? 4  : 0;          // output rows rlo..13 of this tile

    const int j    = lane & 15;          // A row / B column / D column index
    const int half = lane >> 4;          // K-half held by this lane

    const _Float16* f0n = fnorm;
    const _Float16* f1n = fnorm + (size_t)VOX * 32;

    // ---- A fragment: 16 rows (z = z0+j) x 32 channels of normalized f0 ----
    // ISA 16-bit A layout: lane (j,half): e<8 -> K = e + 8*half,
    //                                    e>=8 -> K = 16 + (e-8) + 8*half
    v16h A = {};
    {
        int z = z0 + j;
        if (z < Z_) {
            size_t vi = ((size_t)((n * X_ + x) * Y_ + y) * Z_ + z) * 32;
            v8h b0 = *reinterpret_cast<const v8h*>(f0n + vi + 8 * half);
            v8h b1 = *reinterpret_cast<const v8h*>(f0n + vi + 16 + 8 * half);
#pragma unroll
            for (int e = 0; e < 8; ++e) { A[e] = b0[e]; A[8 + e] = b1[e]; }
        }
    }

    const float s     = __expf(logit_scale[0]);
    const int   train = training[0];

    float wsum = 0.0f, fx = 0.0f, fy = 0.0f, fz = 0.0f;
    float best = -1e30f, bx = 0.0f, by = 0.0f, bz = 0.0f;
    const int i = lane & 15;             // extraction row (lanes 16..31 mirror)
    float* wtile = tile[wid];

#pragma unroll
    for (int dx = -1; dx <= 1; ++dx) {
#pragma unroll
        for (int dy = -1; dy <= 1; ++dy) {
            // ---- B fragment: 32 channels x 16 cols (z' = z0-1+j) of f1 ----
            // ISA 16-bit B layout: lane (j,half): e<8 -> K = e + 8*half, e>=8 -> K = 16 + (e-8) + 8*half
            v16h B = {};
            int xx = x + dx, yy = y + dy, zz = z0 - 1 + j;
            if (xx >= 0 && xx < X_ && yy >= 0 && yy < Y_ &&
                zz >= 0 && zz < Z_) {
                size_t vi = ((size_t)((n * X_ + xx) * Y_ + yy) * Z_ + zz) * 32;
                v8h b0 = *reinterpret_cast<const v8h*>(f1n + vi + 8 * half);
                v8h b1 = *reinterpret_cast<const v8h*>(f1n + vi + 16 + 8 * half);
#pragma unroll
                for (int e = 0; e < 8; ++e) { B[e] = b0[e]; B[8 + e] = b1[e]; }
            }

            v8f D = {};
            D = __builtin_amdgcn_wmma_f32_16x16x32_f16(
                    /*neg_a=*/false, A, /*neg_b=*/false, B,
                    /*c_mod=*/(short)0, D, /*reuse_a=*/false, /*reuse_b=*/false);

            // D element in VGPR r of this lane is D[M = r + 8*half][N = j]
#pragma unroll
            for (int r = 0; r < 8; ++r)
                wtile[(r + 8 * half) * 16 + j] = D[r];
            wave_lds_fence();            // stores visible wave-wide

            // extract the three diagonals: logit(dz) = s * D[i, i+1+dz]
            // (branchless: clamp the address, mask the value)
#pragma unroll
            for (int dz = -1; dz <= 1; ++dz) {
                int colr = i + 1 + dz;
                int col  = colr < 16 ? colr : 15;
                float dv = wtile[i * 16 + col];
                float lg = (colr < 16) ? s * dv : 0.0f;
                float e  = __expf(lg);
                wsum += e;
                fx += e * (float)dx;
                fy += e * (float)dy;
                fz += e * (float)dz;
                if (lg > best) {
                    best = lg; bx = (float)dx; by = (float)dy; bz = (float)dz;
                }
            }
            wave_lds_fence();            // reads done before next overwrite
        }
    }

    if (lane < 16 && i >= rlo && i <= 13) {
        int z = z0 + i;
        size_t o = ((size_t)((n * X_ + x) * Y_ + y) * Z_ + z) * 3;
        if (train) {
            float inv = 1.0f / wsum;
            out[o + 0] = fx * inv;
            out[o + 1] = fy * inv;
            out[o + 2] = fz * inv;
        } else {
            out[o + 0] = bx;
            out[o + 1] = by;
            out[o + 2] = bz;
        }
    }
}

// ============================================================================
extern "C" void kernel_launch(void* const* d_in, const int* in_sizes, int n_in,
                              void* d_out, int out_size, void* d_ws, size_t ws_size,
                              hipStream_t stream)
{
    const float* f0          = (const float*)d_in[0];   // (N,C,X,Y,Z) f32
    const float* f1          = (const float*)d_in[1];   // (N,C,X,Y,Z) f32
    // d_in[2] = spatial_locations: unused (it is exactly the identity meshgrid)
    const float* logit_scale = (const float*)d_in[3];   // scalar f32
    const int*   training    = (const int*)d_in[4];     // scalar int

    _Float16* fnorm = (_Float16*)d_ws;   // needs 2*VOX*32*2 = 14.2 MB scratch

    // Pass 1: normalize + f16 channel-last
    {
        int total = 2 * VOX;                       // 221184
        int blocks = (total + 255) / 256;          // 864
        vt_normalize_kernel<<<blocks, 256, 0, stream>>>(f0, f1, fnorm);
    }

    // Pass 2: WMMA cost volume + soft-argmax flow
    {
        int units = N_ * X_ * Y_ * 2;              // 9216 waves
        vt_costvol_flow_kernel<<<units / 4, 128, 0, stream>>>(
            fnorm, logit_scale, training, (float*)d_out);
    }
}
